// NTXentLoss_73040213835910
// MI455X (gfx1250) — matrix operand order repaired and checked by reading, not compiled, not run on hardware
//
#include <hip/hip_runtime.h>
#include <hip/hip_bf16.h>

// ---------------------------------------------------------------------------
// NT-Xent loss, B=4096, D=512, N=8192, T=0.5
// loss = (1/N) * sum_i [ log( sum_{j != i} exp(sim_ij) ) - sim[i, i^4096] ]
// sim = (zn @ zn^T) * 2, zn = row-normalized concat(z_i, z_j)
// GEMM in split-f16: zn = hi + lo (lo kept unscaled; WMMA preserves denorms),
// sim ~= hi@hi + hi@lo + lo@hi accumulated into ONE f32 WMMA accumulator.
// Wave tile: 64x64 (4x4 grid of 16x16 C tiles) -> 48 WMMAs / 16 fragment
// loads per K-step (~48 FLOP/B register fill). A fragments double-buffered.
// ---------------------------------------------------------------------------

#define BHALF 4096
#define NROWS 8192
#define DDIM  512
#define NSTRIPS 128        // 8192 cols / 64 per strip
#define NROWBLK 16         // 8192 rows / 512 per workgroup
#define NFIN1   32         // finalize stage-1 blocks

typedef _Float16 v16h __attribute__((ext_vector_type(16)));
typedef _Float16 v8h  __attribute__((ext_vector_type(8)));
typedef _Float16 v4h  __attribute__((ext_vector_type(4)));
typedef float    v8f  __attribute__((ext_vector_type(8)));

#define CONCAT16(a, b) __builtin_shufflevector(a, b, 0, 1, 2, 3, 4, 5, 6, 7, \
                                               8, 9, 10, 11, 12, 13, 14, 15)
#define WMMA_F16(A, B, C) __builtin_amdgcn_wmma_f32_16x16x32_f16( \
                              false, A, false, B, (short)0, C, false, false)

// ---------------------------------------------------------------------------
// Phase 1: normalize rows of concat(z_i, z_j); split to f16 hi + lo (exact
// residual, unscaled). One 128-thread block per row, 4 floats per thread.
// ---------------------------------------------------------------------------
__global__ __launch_bounds__(128)
void ntxent_normalize_split(const float* __restrict__ z_i,
                            const float* __restrict__ z_j,
                            _Float16* __restrict__ znh,
                            _Float16* __restrict__ znl) {
    const int r   = blockIdx.x;
    const int tid = threadIdx.x;
    const float* src = (r < BHALF) ? (z_i + (size_t)r * DDIM)
                                   : (z_j + (size_t)(r - BHALF) * DDIM);
    float4 x = ((const float4*)src)[tid];
    float ss = x.x * x.x + x.y * x.y + x.z * x.z + x.w * x.w;
    ss += __shfl_xor(ss, 1, 32);
    ss += __shfl_xor(ss, 2, 32);
    ss += __shfl_xor(ss, 4, 32);
    ss += __shfl_xor(ss, 8, 32);
    ss += __shfl_xor(ss, 16, 32);
    __shared__ float red[4];
    if ((tid & 31) == 0) red[tid >> 5] = ss;
    __syncthreads();
    float total = red[0] + red[1] + red[2] + red[3];
    float inv = 1.0f / fmaxf(sqrtf(total), 1e-8f);

    float zn[4] = { x.x * inv, x.y * inv, x.z * inv, x.w * inv };
    v4h hi, lo;
#pragma unroll
    for (int j = 0; j < 4; ++j) {
        _Float16 h = (_Float16)zn[j];
        hi[j] = h;
        lo[j] = (_Float16)(zn[j] - (float)h);   // denormal-range residual
    }
    size_t base = (size_t)r * DDIM + (size_t)tid * 4;
    *(v4h*)(znh + base) = hi;
    *(v4h*)(znl + base) = lo;
}

// ---------------------------------------------------------------------------
// Phase 2: split-f16 WMMA GEMM tiles + streaming exp-sum reduction.
// grid = (16 row-blocks, 128 column strips), block = 256 threads = 8 waves.
// Each wave owns a 64x64 tile: 4x4 grid of 16x16 WMMA C tiles.
// ---------------------------------------------------------------------------
__global__ __launch_bounds__(256)
void ntxent_sim_logsumexp(const _Float16* __restrict__ znh,
                          const _Float16* __restrict__ znl,
                          float* __restrict__ partials,
                          float* __restrict__ pos) {
    const int lane = threadIdx.x & 31;
    const int wave = threadIdx.x >> 5;
    const int half = lane >> 4;       // 0: lanes 0-15, 1: lanes 16-31
    const int l15  = lane & 15;

    const int rowbase = blockIdx.x * 512 + wave * 64;   // 4 row tiles of 16
    const int colbase = blockIdx.y * 64;                // 4 col tiles of 16

    // Per-lane base pointers; all tile/k displacements are compile-time
    // constants that fold into the load immediate offset.
    const _Float16* pAh = znh + (size_t)(rowbase + l15) * DDIM + half * 8;
    const _Float16* pAl = znl + (size_t)(rowbase + l15) * DDIM + half * 8;
    const _Float16* pBh = znh + (size_t)(colbase + l15) * DDIM + half * 8;
    const _Float16* pBl = znl + (size_t)(colbase + l15) * DDIM + half * 8;

    v8f acc[4][4];
    const v8f zerov = {0.f, 0.f, 0.f, 0.f, 0.f, 0.f, 0.f, 0.f};
#pragma unroll
    for (int rt = 0; rt < 4; ++rt)
#pragma unroll
        for (int ct = 0; ct < 4; ++ct) acc[rt][ct] = zerov;

    // A fragment loader (WMMA 16-bit A 16x32 layout: two 16B chunks/lane).
    auto loadA = [&](int k, v16h* AH, v16h* AL) {
#pragma unroll
        for (int rt = 0; rt < 4; ++rt) {
            const int off = rt * 16 * DDIM + k;
            v8h h0 = *(const v8h*)(pAh + off);
            v8h h1 = *(const v8h*)(pAh + off + 16);
            AH[rt] = CONCAT16(h0, h1);
            v8h l0 = *(const v8h*)(pAl + off);
            v8h l1 = *(const v8h*)(pAl + off + 16);
            AL[rt] = CONCAT16(l0, l1);
        }
    };

    // 12 WMMAs per column tile; B fragment is two 16B reads/lane, same k order as A.
    auto doK = [&](int k, const v16h* AH, const v16h* AL) {
#pragma unroll
        for (int ct = 0; ct < 4; ++ct) {
            const int off = ct * 16 * DDIM + k;
            v16h BH = CONCAT16(*(const v8h*)(pBh + off), *(const v8h*)(pBh + off + 16));
            v16h BL = CONCAT16(*(const v8h*)(pBl + off), *(const v8h*)(pBl + off + 16));
#pragma unroll
            for (int rt = 0; rt < 4; ++rt) {
                acc[rt][ct] = WMMA_F16(AH[rt], BH, acc[rt][ct]);
                acc[rt][ct] = WMMA_F16(AH[rt], BL, acc[rt][ct]);
                acc[rt][ct] = WMMA_F16(AL[rt], BH, acc[rt][ct]);
            }
        }
    };

    // K loop, two 32-wide steps per iteration with double-buffered A frags.
    v16h AH0[4], AL0[4], AH1[4], AL1[4];
    loadA(0, AH0, AL0);
#pragma unroll 1
    for (int k = 0; k < DDIM; k += 64) {
        loadA(k + 32, AH1, AL1);          // prefetch while consuming AH0/AL0
        doK(k, AH0, AL0);
        const int kn = (k + 64 < DDIM) ? (k + 64) : 0;  // last prefetch unused
        loadA(kn, AH0, AL0);
        doK(k + 32, AH1, AL1);
    }

    // Epilogue: C layout -> (row = rowbase + 16*rt + 8*half + v,
    //                        col = colbase + 16*ct + l15)
    float sums[4][8];
#pragma unroll
    for (int rt = 0; rt < 4; ++rt)
#pragma unroll
        for (int v = 0; v < 8; ++v) sums[rt][v] = 0.f;

#pragma unroll
    for (int rt = 0; rt < 4; ++rt) {
        const int rowl = rowbase + rt * 16 + half * 8;
#pragma unroll
        for (int ct = 0; ct < 4; ++ct) {
            const int col = colbase + ct * 16 + l15;
#pragma unroll
            for (int v = 0; v < 8; ++v) {
                const int row = rowl + v;
                const float sim = acc[rt][ct][v] * 2.0f;    // /TEMPERATURE
                if (col == (row ^ BHALF)) pos[row] = sim;   // positive pair
                if (col != row) sums[rt][v] += __expf(sim); // skip diagonal
            }
        }
    }

    // Reduce row sums across each 16-lane half-wave group, write partials.
#pragma unroll
    for (int rt = 0; rt < 4; ++rt) {
#pragma unroll
        for (int v = 0; v < 8; ++v) {
            float s = sums[rt][v];
            s += __shfl_xor(s, 1, 32);
            s += __shfl_xor(s, 2, 32);
            s += __shfl_xor(s, 4, 32);
            s += __shfl_xor(s, 8, 32);
            if (l15 == 0) {
                const int row = rowbase + rt * 16 + half * 8 + v;
                partials[(size_t)blockIdx.y * NROWS + row] = s;
            }
        }
    }
}

// ---------------------------------------------------------------------------
// Phase 3a: per-row combine of strip partials -> per-block loss partials.
// 32 blocks x 256 threads, 256 rows per block, fixed summation order.
// ---------------------------------------------------------------------------
__global__ __launch_bounds__(256)
void ntxent_finalize1(const float* __restrict__ partials,
                      const float* __restrict__ pos,
                      float* __restrict__ blockpart) {
    const int tid = threadIdx.x;
    const int r = blockIdx.x * 256 + tid;
    float S = 0.f;
#pragma unroll 8
    for (int s = 0; s < NSTRIPS; ++s) S += partials[(size_t)s * NROWS + r];
    float val = logf(S) - pos[r];

    __shared__ float red[256];
    red[tid] = val;
    __syncthreads();
    for (int s = 128; s > 0; s >>= 1) {
        if (tid < s) red[tid] += red[tid + s];
        __syncthreads();
    }
    if (tid == 0) blockpart[blockIdx.x] = red[0];
}

// ---------------------------------------------------------------------------
// Phase 3b: final reduce of 32 block partials -> loss. One wave.
// ---------------------------------------------------------------------------
__global__ __launch_bounds__(32)
void ntxent_finalize2(const float* __restrict__ blockpart,
                      float* __restrict__ out) {
    float v = blockpart[threadIdx.x];
    v += __shfl_xor(v, 1, 32);
    v += __shfl_xor(v, 2, 32);
    v += __shfl_xor(v, 4, 32);
    v += __shfl_xor(v, 8, 32);
    v += __shfl_xor(v, 16, 32);
    if (threadIdx.x == 0) out[0] = v / (float)NROWS;
}

// ---------------------------------------------------------------------------
extern "C" void kernel_launch(void* const* d_in, const int* in_sizes, int n_in,
                              void* d_out, int out_size, void* d_ws, size_t ws_size,
                              hipStream_t stream) {
    (void)in_sizes; (void)n_in; (void)out_size; (void)ws_size;
    const float* z_i = (const float*)d_in[0];
    const float* z_j = (const float*)d_in[1];
    float* out = (float*)d_out;

    char* ws = (char*)d_ws;
    const size_t znbytes = (size_t)NROWS * DDIM * sizeof(_Float16); // 8 MB
    _Float16* znh = (_Float16*)ws;
    _Float16* znl = (_Float16*)(ws + znbytes);
    float* partials = (float*)(ws + 2 * znbytes);          // 128*8192 floats
    float* pos = partials + (size_t)NSTRIPS * NROWS;       // 8192 floats
    float* blockpart = pos + NROWS;                        // 32 floats

    ntxent_normalize_split<<<NROWS, 128, 0, stream>>>(z_i, z_j, znh, znl);

    dim3 g2(NROWBLK, NSTRIPS);
    ntxent_sim_logsumexp<<<g2, 256, 0, stream>>>(znh, znl, partials, pos);

    ntxent_finalize1<<<NFIN1, 256, 0, stream>>>(partials, pos, blockpart);
    ntxent_finalize2<<<1, 32, 0, stream>>>(blockpart, out);
}
